// FullCalibratedModel_56109452755472
// MI455X (gfx1250) — matrix operand order repaired and checked by reading, not compiled, not run on hardware
//
#include <hip/hip_runtime.h>
#include <hip/hip_bf16.h>

typedef __attribute__((ext_vector_type(16))) _Float16 v16h;
typedef __attribute__((ext_vector_type(8)))  float    v8f;

union AFrag { v16h v; _Float16 e[16]; };
union CFrag { v8f  v; float    e[8];  };

// ---- WMMA helpers (CDNA5: V_WMMA_F32_16X16X32_F16, D = A(16x32) * B(32x16) + C) ----

__device__ __forceinline__ v8f wmma_f16(v16h a, v16h b, v8f c) {
  // 8 args: (neg_a, A, neg_b, B, c_mod, C, reuse_a, reuse_b)
  return __builtin_amdgcn_wmma_f32_16x16x32_f16(false, a, false, b, (short)0, c,
                                                false, false);
}

// A-matrix fragment: source row-major [M][K] (f16). ISA 7.12.2 16-bit A 16x32:
// lanes 0-15: M=lane, VGPR0..3 hold K=0..7 (pairs), VGPR4..7 hold K=16..23;
// lanes 16-31 same rows but K offset +8 in each group.
__device__ __forceinline__ v16h load_a_frag(const _Float16* __restrict__ A,
                                            int lda, int lane, int kbase) {
  const int m  = lane & 15;
  const int kh = (lane >> 4) << 3;   // 0 or 8
  AFrag f;
#pragma unroll
  for (int idx = 0; idx < 16; ++idx) {
    const int v = idx >> 1, p = idx & 1;
    const int k = kbase + ((v < 4) ? 0 : 16) + kh + ((v & 3) << 1) + p;
    f.e[idx] = A[(size_t)m * lda + k];
  }
  return f.v;
}

// B-matrix fragment where B[k][n] = W[n][k] with W row-major [N][K] (f16).
// lanes 0-15: N=lane, K=kbase+0..7,16..23; lanes 16-31: N=lane-16, K=kbase+8..15,24..31.
__device__ __forceinline__ v16h load_b_frag(const _Float16* __restrict__ W,
                                            int ldw, int n0, int lane, int kbase) {
  const int n  = n0 + (lane & 15);
  const int kh = (lane >> 4) << 3;   // 0 or 8
  AFrag f;
#pragma unroll
  for (int idx = 0; idx < 16; ++idx)
    f.e[idx] = W[(size_t)n * ldw + kbase + kh + idx + ((idx < 8) ? 0 : 8)];
  return f.v;
}

// ---- K0: f32 -> f16 weight conversion ----
__global__ void k0_cvt(const float* __restrict__ src, _Float16* __restrict__ dst, int n) {
  int i = blockIdx.x * blockDim.x + threadIdx.x;
  if (i < n) dst[i] = (_Float16)src[i];
}

// ---- K1: embedding gather + FFN (two WMMA GEMMs) + residual + LayerNorm ----
// One wave per 16-token tile. 2 waves / block. LDS rows padded 16B so the
// ds_load_b128 A-fragment reads hit 64 distinct banks (stride 68 / 132 words).
__global__ __launch_bounds__(64) void k1_embed_ffn_ln(
    const int* __restrict__ seq, const float* __restrict__ embed,
    const _Float16* __restrict__ W1h, const float* __restrict__ b1,
    const _Float16* __restrict__ W2h, const float* __restrict__ b2,
    const float* __restrict__ gamma, const float* __restrict__ beta,
    _Float16* __restrict__ hbuf)
{
  __shared__ _Float16 sh_h[2][16][136];    // 128 + 8 pad halves
  __shared__ _Float16 sh_act[2][16][264];  // 256 + 8 pad halves
  __shared__ float    sh_x[2][16][132];    // 128 + 4 pad floats
  __shared__ int      sh_row[2][16];

  const int wave = threadIdx.x >> 5;
  const int lane = threadIdx.x & 31;
  const int t0   = (blockIdx.x * 2 + wave) * 16;   // first flat token of tile

  if (lane < 16) sh_row[wave][lane] = seq[t0 + lane];
  __syncthreads();

  for (int e = lane; e < 16 * 128; e += 32) {
    const int m = e >> 7, k = e & 127;
    sh_h[wave][m][k] = (_Float16)embed[(size_t)sh_row[wave][m] * 128 + k];
  }
  __syncthreads();

  v16h afrag[4];
#pragma unroll
  for (int kc = 0; kc < 4; ++kc)
    afrag[kc] = load_a_frag(&sh_h[wave][0][0], 136, lane, kc * 32);

  const int n_lo  = lane & 15;
  const int mbase = (lane >> 4) << 3;

  // matmul1: (16x128) @ W1^T (128x256), +b1, relu -> sh_act (f16)
  for (int nt = 0; nt < 16; ++nt) {
    const int n0 = nt * 16, n = n0 + n_lo;
    v8f zero = {};
    CFrag acc; acc.v = zero;
#pragma unroll
    for (int kc = 0; kc < 4; ++kc)
      acc.v = wmma_f16(afrag[kc], load_b_frag(W1h, 128, n0, lane, kc * 32), acc.v);
    const float bn = b1[n];
#pragma unroll
    for (int r = 0; r < 8; ++r)
      sh_act[wave][mbase + r][n] = (_Float16)fmaxf(acc.e[r] + bn, 0.0f);
  }
  __syncthreads();

  // matmul2: (16x256) @ W2^T (256x128), +b2, +residual -> sh_x (f32)
  for (int nt = 0; nt < 8; ++nt) {
    const int n0 = nt * 16, n = n0 + n_lo;
    v8f zero = {};
    CFrag acc; acc.v = zero;
#pragma unroll
    for (int kc = 0; kc < 8; ++kc) {
      v16h a = load_a_frag(&sh_act[wave][0][0], 264, lane, kc * 32);
      acc.v = wmma_f16(a, load_b_frag(W2h, 256, n0, lane, kc * 32), acc.v);
    }
    const float bn = b2[n];
#pragma unroll
    for (int r = 0; r < 8; ++r) {
      const int m = mbase + r;
      sh_x[wave][m][n] = acc.e[r] + bn + (float)sh_h[wave][m][n];
    }
  }
  __syncthreads();

  // LayerNorm: 2 lanes per token (each sums 64 of the 128 channels)
  {
    const int m  = lane >> 1;
    const int kb = (lane & 1) << 6;
    float s1 = 0.0f, s2 = 0.0f;
    for (int k = 0; k < 64; ++k) {
      const float x = sh_x[wave][m][kb + k];
      s1 += x; s2 += x * x;
    }
    s1 += __shfl_xor(s1, 1, 32);
    s2 += __shfl_xor(s2, 1, 32);
    const float mu   = s1 * (1.0f / 128.0f);
    const float var  = s2 * (1.0f / 128.0f) - mu * mu;
    const float rstd = rsqrtf(var + 1e-5f);
    const size_t tok = (size_t)t0 + m;
    for (int k = 0; k < 64; ++k) {
      const int kk = kb + k;
      const float x = sh_x[wave][m][kk];
      hbuf[tok * 128 + kk] = (_Float16)((x - mu) * rstd * gamma[kk] + beta[kk]);
    }
  }
}

// ---- K2: ks_all/ke_all = h[:, :-1] @ Wsem^T / Wepi^T (WMMA) ----
// One wave per (batch, 16-token tile); stores [t][b][c] layout for the scan.
__global__ __launch_bounds__(128) void k2_keys(
    const _Float16* __restrict__ hbuf,
    const _Float16* __restrict__ Wsemh, const _Float16* __restrict__ Wepih,
    float* __restrict__ ksall, float* __restrict__ keall)
{
  const int gw   = (blockIdx.x * blockDim.x + threadIdx.x) >> 5;
  const int lane = threadIdx.x & 31;
  const int b  = gw >> 6;      // 0..63
  const int t0 = (gw & 63) * 16;

  v16h afrag[4];
#pragma unroll
  for (int kc = 0; kc < 4; ++kc)
    afrag[kc] = load_a_frag(hbuf + ((size_t)b * 1024 + t0) * 128, 128, lane, kc * 32);

  const int mbase = (lane >> 4) << 3;
  const int n_lo  = lane & 15;
#pragma unroll
  for (int sel = 0; sel < 2; ++sel) {
    const _Float16* W = sel ? Wepih : Wsemh;
    float* out = sel ? keall : ksall;
    for (int nt = 0; nt < 4; ++nt) {
      const int n0 = nt * 16;
      v8f zero = {};
      CFrag acc; acc.v = zero;
#pragma unroll
      for (int kc = 0; kc < 4; ++kc)
        acc.v = wmma_f16(afrag[kc], load_b_frag(W, 128, n0, lane, kc * 32), acc.v);
#pragma unroll
      for (int r = 0; r < 8; ++r) {
        const int t = t0 + mbase + r;
        if (t < 1023) out[((size_t)t * 64 + b) * 64 + n0 + n_lo] = acc.e[r];
      }
    }
  }
}

// ---- K3: sequential fast-weight scan. One workgroup per batch; M_s/M_e in LDS.
// 4 barriers/step; Ms/Me rows padded to 65 floats (conflict-free banks);
// key vectors double-buffered in registers (loads for t+1 issued during step t).
__global__ __launch_bounds__(256) void k3_scan(
    const float* __restrict__ ksall, const float* __restrict__ keall,
    float* __restrict__ Msg, float* __restrict__ Meg)
{
  __shared__ float Ms[64][65];
  __shared__ float Me[64][65];
  __shared__ float ksv[64], kev[64], knsv[64], knev[64], dsv[64], dev[64];
  __shared__ float sn[4];   // ||ks||, ||ke||, ||ds||, ||de||

  const int b    = blockIdx.x;
  const int tid  = threadIdx.x;
  const int lane = tid & 31;
  const int wv   = tid >> 5;

  for (int e = tid; e < 4096; e += 256) { Ms[e >> 6][e & 63] = 0.0f; Me[e >> 6][e & 63] = 0.0f; }

  // preload step 0 key vectors: wave0 owns ks, wave1 owns ke (2 floats/lane)
  float ra = 0.0f, rb = 0.0f;
  if (wv == 0) {
    ra = ksall[(size_t)b * 64 + lane];
    rb = ksall[(size_t)b * 64 + lane + 32];
  } else if (wv == 1) {
    ra = keall[(size_t)b * 64 + lane];
    rb = keall[(size_t)b * 64 + lane + 32];
  }

  const int i  = tid >> 2;         // output row (4 threads per row)
  const int jb = (tid & 3) << 4;   // 16-wide j slice
  __syncthreads();

  for (int t = 0; t < 1023; ++t) {
    // phase B: norm + normalize fully in-wave; publish to LDS; start next loads
    if (wv < 2) {
      float v = ra * ra + rb * rb;
#pragma unroll
      for (int off = 16; off; off >>= 1) v += __shfl_xor(v, off, 32);
      const float nrm = sqrtf(v);
      const float rs  = 1.0f / fmaxf(nrm, 1e-12f);
      float* kv = (wv == 0) ? ksv : kev;
      float* kn = (wv == 0) ? knsv : knev;
      kv[lane]      = ra;
      kv[lane + 32] = rb;
      kn[lane]      = ra * rs;
      kn[lane + 32] = rb * rs;
      if (lane == 0) sn[wv] = nrm;
      if (t + 1 < 1023) {   // double-buffered load: waited on one iter later
        const size_t nb = ((size_t)(t + 1) * 64 + b) * 64;
        const float* src = (wv == 0) ? ksall : keall;
        ra = src[nb + lane];
        rb = src[nb + lane + 32];
      }
    } else if (wv == 2 && lane < 2 && t + 2 < 1023) {
      const size_t nb2 = ((size_t)(t + 2) * 64 + b) * 64;
      __builtin_prefetch(&ksall[nb2 + lane * 32], 0, 0);  // -> global_prefetch_b8
      __builtin_prefetch(&keall[nb2 + lane * 32], 0, 0);
    }
    __syncthreads();

    // phase C: vps = M_s @ kns, vpe = M_e @ kne  (bank-conflict-free, padded)
    {
      float ps = 0.0f, pe = 0.0f;
#pragma unroll
      for (int jj = 0; jj < 16; ++jj) {
        ps += Ms[i][jb + jj] * knsv[jb + jj];
        pe += Me[i][jb + jj] * knev[jb + jj];
      }
      ps += __shfl_xor(ps, 1, 32); ps += __shfl_xor(ps, 2, 32);
      pe += __shfl_xor(pe, 1, 32); pe += __shfl_xor(pe, 2, 32);
      if ((tid & 3) == 0) { dsv[i] = ksv[i] - ps; dev[i] = kev[i] - pe; }
    }
    __syncthreads();

    // phase D: error norms
    if (wv < 2) {
      const float* dv = (wv == 0) ? dsv : dev;
      float v = dv[lane] * dv[lane] + dv[lane + 32] * dv[lane + 32];
#pragma unroll
      for (int off = 16; off; off >>= 1) v += __shfl_xor(v, off, 32);
      if (lane == 0) sn[2 + wv] = sqrtf(v);
    }
    __syncthreads();

    // phase E: gated rank-1 update
    if ((sn[2] >= 0.7f * sn[0]) || (sn[3] >= 0.7f * sn[1])) {
      const float wt = (float)(t + 1) * (1.0f / 1024.0f);
      const float gs = 0.05f * dsv[i];
      const float ge = 0.05f * wt * dev[i];
#pragma unroll
      for (int jj = 0; jj < 16; ++jj) {
        Ms[i][jb + jj] += gs * knsv[jb + jj];
        Me[i][jb + jj] += ge * knev[jb + jj];
      }
    }
    __syncthreads();
  }

  for (int e = tid; e < 4096; e += 256) {
    Msg[(size_t)b * 4096 + e] = Ms[e >> 6][e & 63];
    Meg[(size_t)b * 4096 + e] = Me[e >> 6][e & 63];
  }
}

// ---- K4a: qs/qe = h[:, -1] @ Wsem^T / Wepi^T ----
__global__ void k4a_q(const _Float16* __restrict__ hbuf,
                      const _Float16* __restrict__ Wsemh, const _Float16* __restrict__ Wepih,
                      float* __restrict__ qs, float* __restrict__ qe)
{
  const int tid = blockIdx.x * blockDim.x + threadIdx.x;   // 0..8191
  const int sel = tid >> 12, idx = tid & 4095;
  const int b = idx >> 6, c = idx & 63;
  const _Float16* W = sel ? Wepih : Wsemh;
  const _Float16* h = hbuf + ((size_t)b * 1024 + 1023) * 128;
  float s = 0.0f;
  for (int k = 0; k < 128; ++k) s += (float)h[k] * (float)W[(size_t)c * 128 + k];
  (sel ? qe : qs)[idx] = s;
}

// ---- K4b: cs/ce = M_s @ qs, M_e @ qe ----
__global__ void k4b_ctx(const float* __restrict__ Msg, const float* __restrict__ Meg,
                        const float* __restrict__ qs, const float* __restrict__ qe,
                        float* __restrict__ cs, float* __restrict__ ce)
{
  const int tid = blockIdx.x * blockDim.x + threadIdx.x;   // 0..8191
  const int sel = tid >> 12, idx = tid & 4095;
  const int b = idx >> 6, i = idx & 63;
  const float* M = (sel ? Meg : Msg) + (size_t)b * 4096 + (size_t)i * 64;
  const float* q = (sel ? qe : qs) + b * 64;
  float s = 0.0f;
  for (int j = 0; j < 64; ++j) s += M[j] * q[j];
  (sel ? ce : cs)[idx] = s;
}

// ---- K4c: r = concat(cs,ce) @ Wrp^T + brp, stored f16 for the WMMA vocab GEMM ----
__global__ void k4c_rp(const float* __restrict__ cs, const float* __restrict__ ce,
                       const float* __restrict__ Wrp, const float* __restrict__ brp,
                       _Float16* __restrict__ rh)
{
  const int tid = blockIdx.x * blockDim.x + threadIdx.x;   // 0..8191
  const int b = tid >> 7, i = tid & 127;
  const float* w = Wrp + (size_t)i * 128;
  float s = brp[i];
  for (int j = 0; j < 64; ++j) s += cs[b * 64 + j] * w[j];
  for (int j = 0; j < 64; ++j) s += ce[b * 64 + j] * w[64 + j];
  rh[(size_t)b * 128 + i] = (_Float16)s;
}

// ---- K4d: out = r @ Wout^T + bout (64x32000, K=128) via WMMA ----
__global__ __launch_bounds__(128) void k4d_out(
    const _Float16* __restrict__ rh, const _Float16* __restrict__ Wouth,
    const float* __restrict__ bout, float* __restrict__ out)
{
  const int gw   = (blockIdx.x * blockDim.x + threadIdx.x) >> 5;  // 0..7999
  const int lane = threadIdx.x & 31;
  const int m0 = (gw & 3) * 16;
  const int n0 = (gw >> 2) * 16;

  v16h afrag[4];
#pragma unroll
  for (int kc = 0; kc < 4; ++kc)
    afrag[kc] = load_a_frag(rh + (size_t)m0 * 128, 128, lane, kc * 32);

  v8f zero = {};
  CFrag acc; acc.v = zero;
#pragma unroll
  for (int kc = 0; kc < 4; ++kc)
    acc.v = wmma_f16(afrag[kc], load_b_frag(Wouth, 128, n0, lane, kc * 32), acc.v);

  const int n = n0 + (lane & 15);
  const float bn = bout[n];
  const int mbase = (lane >> 4) << 3;
#pragma unroll
  for (int r = 0; r < 8; ++r)
    out[(size_t)(m0 + mbase + r) * 32000 + n] = acc.e[r] + bn;
}

extern "C" void kernel_launch(void* const* d_in, const int* in_sizes, int n_in,
                              void* d_out, int out_size, void* d_ws, size_t ws_size,
                              hipStream_t stream)
{
  (void)in_sizes; (void)n_in; (void)out_size; (void)ws_size;

  const int*   seq   = (const int*)  d_in[0];
  const float* embed = (const float*)d_in[1];
  const float* W1    = (const float*)d_in[2];
  const float* b1    = (const float*)d_in[3];
  const float* W2    = (const float*)d_in[4];
  const float* b2    = (const float*)d_in[5];
  const float* gamma = (const float*)d_in[6];
  const float* beta  = (const float*)d_in[7];
  const float* Wsem  = (const float*)d_in[8];
  const float* Wepi  = (const float*)d_in[9];
  const float* Wrp   = (const float*)d_in[10];
  const float* brp   = (const float*)d_in[11];
  const float* Wout  = (const float*)d_in[12];
  const float* bout  = (const float*)d_in[13];
  float* out = (float*)d_out;

  char* ws = (char*)d_ws;
  size_t off = 0;
  auto carve = [&](size_t bytes) -> void* {
    void* p = ws + off;
    off = (off + bytes + 255) & ~(size_t)255;
    return p;
  };

  _Float16* hbuf  = (_Float16*)carve((size_t)64 * 1024 * 128 * 2);   // 16.8 MB
  float*    ksall = (float*)   carve((size_t)1023 * 64 * 64 * 4);    // 16.8 MB
  float*    keall = (float*)   carve((size_t)1023 * 64 * 64 * 4);    // 16.8 MB
  float*    Msg   = (float*)   carve((size_t)64 * 64 * 64 * 4);      // 1 MB
  float*    Meg   = (float*)   carve((size_t)64 * 64 * 64 * 4);      // 1 MB
  _Float16* W1h   = (_Float16*)carve(256 * 128 * 2);
  _Float16* W2h   = (_Float16*)carve(128 * 256 * 2);
  _Float16* Wsemh = (_Float16*)carve(64 * 128 * 2);
  _Float16* Wepih = (_Float16*)carve(64 * 128 * 2);
  _Float16* Wouth = (_Float16*)carve((size_t)32000 * 128 * 2);       // 8.2 MB
  float*    qs    = (float*)   carve(64 * 64 * 4);
  float*    qe    = (float*)   carve(64 * 64 * 4);
  float*    cs    = (float*)   carve(64 * 64 * 4);
  float*    ce    = (float*)   carve(64 * 64 * 4);
  _Float16* rh    = (_Float16*)carve(64 * 128 * 2);

  // K0: weight conversions to f16
  k0_cvt<<<(256 * 128 + 255) / 256, 256, 0, stream>>>(W1, W1h, 256 * 128);
  k0_cvt<<<(128 * 256 + 255) / 256, 256, 0, stream>>>(W2, W2h, 128 * 256);
  k0_cvt<<<(64 * 128 + 255) / 256, 256, 0, stream>>>(Wsem, Wsemh, 64 * 128);
  k0_cvt<<<(64 * 128 + 255) / 256, 256, 0, stream>>>(Wepi, Wepih, 64 * 128);
  k0_cvt<<<(32000 * 128 + 255) / 256, 256, 0, stream>>>(Wout, Wouth, 32000 * 128);

  // K1: embed + FFN + LN  (65536 tokens / 16 per wave / 2 waves per block)
  k1_embed_ffn_ln<<<2048, 64, 0, stream>>>(seq, embed, W1h, b1, W2h, b2,
                                           gamma, beta, hbuf);

  // K2: key projections (64 batches x 64 t-tiles = 4096 waves, 4 waves/block)
  k2_keys<<<1024, 128, 0, stream>>>(hbuf, Wsemh, Wepih, ksall, keall);

  // K3: sequential scan, one workgroup per batch element
  k3_scan<<<64, 256, 0, stream>>>(ksall, keall, Msg, Meg);

  // K4: readout chain
  k4a_q<<<32, 256, 0, stream>>>(hbuf, Wsemh, Wepih, qs, qe);
  k4b_ctx<<<32, 256, 0, stream>>>(Msg, Meg, qs, qe, cs, ce);
  k4c_rp<<<32, 256, 0, stream>>>(cs, ce, Wrp, brp, rh);
  k4d_out<<<2000, 128, 0, stream>>>(rh, Wouth, bout, out);
}
